// SkipGram_32530082300266
// MI455X (gfx1250) — matrix operand order repaired and checked by reading, not compiled, not run on hardware
//
#include <hip/hip_runtime.h>
#include <hip/hip_bf16.h>

// SkipGram negative-sampling loss for MI455X (gfx1250, wave32).
// Memory-bound gather workload (~184 MB @ 23.3 TB/s ~ 8 us floor); dot
// products routed through v_wmma_f32_16x16x32_f16 (B-matrix = u replicated
// across all 16 columns, A-matrix = [v_e ; neg rows] tiles).

typedef __attribute__((ext_vector_type(16))) _Float16 v16h;
typedef __attribute__((ext_vector_type(8)))  float    v8f;

#define EMB   128
#define KNEG  20

__device__ __forceinline__ float log_sigmoid(float x) {
    // logσ(x) = min(x,0) - log1p(exp(-|x|))  (numerically stable)
    return fminf(x, 0.0f) - log1pf(expf(-fabsf(x)));
}

// One wave32 per row b. 4 waves per block.
__global__ void __launch_bounds__(128)
sgns_wmma_kernel(const int* __restrict__ u_idx,
                 const int* __restrict__ v_idx,
                 const int* __restrict__ neg_idx,
                 const float* __restrict__ u_emb,
                 const float* __restrict__ v_emb,
                 float* __restrict__ partial,
                 int Btot) {
    const int lane = threadIdx.x & 31;
    const int wave = threadIdx.x >> 5;
    const int b    = blockIdx.x * 4 + wave;     // wave-uniform
    if (b >= Btot) return;                      // uniform per wave -> EXEC stays all-1s

    const int  r  = lane & 15;                  // tile row handled by this lane
    const bool hi = (lane >= 16);

    // ---- gather indices -------------------------------------------------
    const long long bu = u_idx[b];
    const long long bv = v_idx[b];
    const float* __restrict__ uptr = u_emb + bu * EMB;

    // Tile0 rows: row0 = v_e, rows 1..15 = neg 0..14
    long long i0 = (r == 0) ? bv : (long long)neg_idx[b * KNEG + (r - 1)];
    // Tile1 rows: rows 16..20 = neg 15..19, rows 21..31 clamped (ignored later)
    int r2 = (r < 5) ? r : 4;
    long long i1 = (long long)neg_idx[b * KNEG + 15 + r2];
    const float* __restrict__ row0 = v_emb + i0 * EMB;
    const float* __restrict__ row1 = v_emb + i1 * EMB;

    // ---- 4 K-chunks of 32, two row-tiles, accumulate in f32 -------------
    v8f acc0 = {};   // rows 0..15
    v8f acc1 = {};   // rows 16..31 (only 16..20 meaningful)
#pragma unroll
    for (int c = 0; c < 4; ++c) {
        const int base = c * 32;
        // ISA 7.12.2: 16-bit A 16x32 — lanes 0-15: K {0-7,16-23}; lanes 16-31: K {8-15,24-31}
        const int klo = base + (hi ? 8 : 0);
        // 16-bit B 32x16 — lanes 0-15: K {0-7,16-23}; lanes 16-31: K {8-15,24-31} (col = lane%16)
        const int kb  = base + (hi ? 8 : 0);

        v16h a0, a1, bf;
#pragma unroll
        for (int e = 0; e < 8; ++e) {
            a0[e]     = (_Float16)row0[klo + e];
            a0[e + 8] = (_Float16)row0[klo + 16 + e];
            a1[e]     = (_Float16)row1[klo + e];
            a1[e + 8] = (_Float16)row1[klo + 16 + e];
        }
#pragma unroll
        for (int e = 0; e < 16; ++e)
            bf[e] = (_Float16)uptr[kb + e + (e & 8)];

        acc0 = __builtin_amdgcn_wmma_f32_16x16x32_f16(false, a0, false, bf,
                                                      (short)0, acc0, false, false);
        acc1 = __builtin_amdgcn_wmma_f32_16x16x32_f16(false, a1, false, bf,
                                                      (short)0, acc1, false, false);
    }

    // ---- extract dots ---------------------------------------------------
    // D layout: lanes 0-15, VGPR e -> row e ; lanes 16-31, VGPR e -> row 8+e.
    // All 16 columns identical (B columns replicated), so any column works.
    float s  = acc0[0];                    // row 0 = u.v        (valid lanes 0-15)
    float pa = 0.0f, pb = 0.0f;
#pragma unroll
    for (int e = 1; e < 8; ++e) pa += acc0[e];   // rows 1..7     (lanes 0-15)
#pragma unroll
    for (int e = 0; e < 5; ++e) pa += acc1[e];   // rows 16..20   (lanes 0-15)
#pragma unroll
    for (int e = 0; e < 8; ++e) pb += acc0[e];   // rows 8..15    (lanes 16-31)

    float ns   = pa + __shfl(pb, 16, 32);        // full negative score (lanes 0-15)
    float loss = -(log_sigmoid(s) + log_sigmoid(-ns));

    // ---- block partial (deterministic) ----------------------------------
    __shared__ float wsum[4];
    if (lane == 0) wsum[wave] = loss;
    __syncthreads();
    if (threadIdx.x == 0)
        partial[blockIdx.x] = (wsum[0] + wsum[1]) + (wsum[2] + wsum[3]);
}

// Deterministic fixed-tree reduction of block partials -> scalar mean loss.
__global__ void __launch_bounds__(256)
sgns_reduce_kernel(const float* __restrict__ partial, float* __restrict__ out,
                   int n, float scale) {
    __shared__ float sm[256];
    float s = 0.0f;
    for (int i = threadIdx.x; i < n; i += 256) s += partial[i];
    sm[threadIdx.x] = s;
    __syncthreads();
    for (int off = 128; off > 0; off >>= 1) {
        if ((int)threadIdx.x < off) sm[threadIdx.x] += sm[threadIdx.x + off];
        __syncthreads();
    }
    if (threadIdx.x == 0) out[0] = sm[0] * scale;
}

extern "C" void kernel_launch(void* const* d_in, const int* in_sizes, int n_in,
                              void* d_out, int out_size, void* d_ws, size_t ws_size,
                              hipStream_t stream) {
    const int*   u    = (const int*)d_in[0];    // [B]
    const int*   v    = (const int*)d_in[1];    // [B]
    const int*   negv = (const int*)d_in[2];    // [B,K]
    const float* uemb = (const float*)d_in[3];  // [VOCAB,128]
    const float* vemb = (const float*)d_in[4];  // [VOCAB,128]
    float*       out  = (float*)d_out;
    float*       part = (float*)d_ws;

    const int B       = in_sizes[0];            // 16384
    const int nblocks = (B + 3) / 4;            // 4 rows (waves) per block

    sgns_wmma_kernel<<<nblocks, 128, 0, stream>>>(u, v, negv, uemb, vemb, part, B);
    sgns_reduce_kernel<<<1, 256, 0, stream>>>(part, out, nblocks, 1.0f / (float)B);
}
